// BahdanauAttentionAudio_10754598109813
// MI455X (gfx1250) — matrix operand order repaired and checked by reading, not compiled, not run on hardware
//
#include <hip/hip_runtime.h>
#include <hip/hip_bf16.h>
#include <math.h>

#define B_   128
#define T_   1998
#define H_   256
#define U_   256
#define K_   1332
#define MTOT (B_ * T_)          // 255744 = 1998 * 128 exactly

typedef __attribute__((ext_vector_type(16))) __bf16 v16bf;
typedef __attribute__((ext_vector_type(8)))  float  v8f;

union Frag32 { uint4 u[2]; v16bf v; };
union Pack4  { __bf16 h[4]; uint2 u2; };

__device__ __forceinline__ float fast_exp(float x) {
    return __builtin_amdgcn_exp2f(x * 1.44269504088896340736f);
}
__device__ __forceinline__ float fast_tanh(float x) {
    // 1 - 2/(e^{2x}+1); exp2->inf => +1, exp2->0 => -1 (branch-free saturation)
    return 1.0f - 2.0f * __builtin_amdgcn_rcpf(fast_exp(2.0f * x) + 1.0f);
}
__device__ __forceinline__ float fast_sigmoid(float x) {
    return __builtin_amdgcn_rcpf(1.0f + fast_exp(-x));
}

// ---------------------------------------------------------------- k1: q_proj
__global__ void qproj_kernel(const float* __restrict__ query,
                             const float* __restrict__ W2w,
                             const float* __restrict__ W2b,
                             float* __restrict__ qproj) {
    __shared__ __align__(16) float q[U_];
    const int b = blockIdx.x, n = threadIdx.x;
    q[n] = query[b * H_ + n];
    __syncthreads();
    const float4* w4 = reinterpret_cast<const float4*>(W2w + (size_t)n * H_);
    const float4* q4 = reinterpret_cast<const float4*>(q);
    float acc = 0.f;
#pragma unroll 4
    for (int k = 0; k < H_ / 4; k++) {
        float4 wv = w4[k], qv = q4[k];
        acc += wv.x * qv.x + wv.y * qv.y + wv.z * qv.z + wv.w * qv.w;
    }
    qproj[b * U_ + n] = acc + W2b[n];
}

// ------------------------------------------------- k2a: prev_att -> [T,B]
__global__ void transpose_pa(const float* __restrict__ pa, float* __restrict__ paT) {
    int gid = blockIdx.x * 256 + threadIdx.x;
    if (gid < MTOT) {
        int b = gid / T_;
        int t = gid - b * T_;
        paT[t * B_ + b] = pa[gid];
    }
}

// ----------------------- k2b: convoT[t][b] = sum_i paT[i][b] * conv_w[t][i][3]
__global__ void conv_kernel(const float* __restrict__ paT,
                            const float* __restrict__ convw,
                            float* __restrict__ convoT) {
    const int t0 = blockIdx.x * 16;
    const int b  = threadIdx.x;          // 128 threads
    const int tmax = min(16, T_ - t0);
    float acc[16];
#pragma unroll
    for (int j = 0; j < 16; j++) acc[j] = 0.f;
    for (int i = 0; i < T_; i++) {
        float p = paT[i * B_ + b];       // coalesced, L2-resident (reused 16x)
#pragma unroll
        for (int j = 0; j < 16; j++) {
            if (j < tmax) {
                // wave-uniform address -> scalar load
                float w = convw[((size_t)(t0 + j) * T_ + i) * 7 + 3];
                acc[j] += p * w;
            }
        }
    }
    for (int j = 0; j < tmax; j++) convoT[(t0 + j) * B_ + b] = acc[j];
}

// ------------- k3: fused bf16-WMMA GEMM + bias + loc + tanh + V reduction
__global__ void score_gemm_kernel(const float* __restrict__ values,
                                  const float* __restrict__ W1w,
                                  const float* __restrict__ W1b,
                                  const float* __restrict__ qproj,
                                  const float* __restrict__ convoT,
                                  const float* __restrict__ locproj,
                                  const float* __restrict__ Vw,
                                  const float* __restrict__ Vb,
                                  float* __restrict__ rawScore) {
    extern __shared__ char smem[];
    const int PITCH = 264;                           // halves; 528B rows (8/16B mult.)
    __bf16* sW1 = (__bf16*)smem;                     // [256][264] = 135168 B
    __bf16* sA  = (__bf16*)(smem + 256 * PITCH * 2); // [128][264] =  67584 B

    const int tid = threadIdx.x;
    const int rowbase = blockIdx.x * 128;

    // stage W1 (f32 -> bf16), float4 loads, packed b64 LDS stores
    const float4* w14 = reinterpret_cast<const float4*>(W1w);
    for (int q = tid; q < U_ * H_ / 4; q += 256) {
        float4 f = w14[q];
        int n = q >> 6, kk = (q & 63) << 2;
        Pack4 p;
        p.h[0] = (__bf16)f.x; p.h[1] = (__bf16)f.y;
        p.h[2] = (__bf16)f.z; p.h[3] = (__bf16)f.w;
        *reinterpret_cast<uint2*>(sW1 + n * PITCH + kk) = p.u2;
    }
    // stage A tile (128 rows of values), same pattern
    const float4* v4 = reinterpret_cast<const float4*>(values + (size_t)rowbase * H_);
    for (int q = tid; q < 128 * 64; q += 256) {
        float4 f = v4[q];
        int r = q >> 6, kk = (q & 63) << 2;
        Pack4 p;
        p.h[0] = (__bf16)f.x; p.h[1] = (__bf16)f.y;
        p.h[2] = (__bf16)f.z; p.h[3] = (__bf16)f.w;
        *reinterpret_cast<uint2*>(sA + r * PITCH + kk) = p.u2;
    }
    __syncthreads();

    const int wave = tid >> 5;
    const int lane = tid & 31;
    const int lr   = lane & 15;   // A row-in-tile / B-C column
    const int kh   = lane >> 4;

    // all 8 A fragments (K = 0..255) held in VGPRs, reused across all N-tiles.
    // ISA 7.12.2 16-bit A layout: lane<16 holds K in {kh*8..+7} u {16+kh*8..+7}
    Frag32 afrag[8];
    const __bf16* aRow = sA + (wave * 16 + lr) * PITCH;
#pragma unroll
    for (int s = 0; s < 8; s++) {
        afrag[s].u[0] = *reinterpret_cast<const uint4*>(aRow + s * 32 + kh * 8);
        afrag[s].u[1] = *reinterpret_cast<const uint4*>(aRow + s * 32 + 16 + kh * 8);
    }

    // per-row metadata (C/D layout: element idx -> row idx + 8*kh)
    int mrow[8]; float cvv[8]; int qb[8];
#pragma unroll
    for (int idx = 0; idx < 8; idx++) {
        int m = rowbase + wave * 16 + idx + 8 * kh;
        mrow[idx] = m;
        int bb = m / T_;
        int tt = m - bb * T_;
        cvv[idx] = convoT[tt * B_ + bb];
        qb[idx]  = bb << 8;
    }

    float sacc[8] = {0, 0, 0, 0, 0, 0, 0, 0};

    // two N-tiles per iteration: two independent accumulators keep the
    // matrix pipe busy (wmma pairs have no RAW dependence on each other)
    for (int j = 0; j < 16; j += 2) {
        v8f acc0 = {}, acc1 = {};
        const __bf16* bRow0 = sW1 + (j * 16 + lr) * PITCH + kh * 8;
        const __bf16* bRow1 = bRow0 + 16 * PITCH;
#pragma unroll
        for (int s = 0; s < 8; s++) {               // K steps of 32
            Frag32 b0, b1;
            b0.u[0] = *reinterpret_cast<const uint4*>(bRow0 + s * 32);
            b0.u[1] = *reinterpret_cast<const uint4*>(bRow0 + s * 32 + 16);
            b1.u[0] = *reinterpret_cast<const uint4*>(bRow1 + s * 32);
            b1.u[1] = *reinterpret_cast<const uint4*>(bRow1 + s * 32 + 16);
            acc0 = __builtin_amdgcn_wmma_f32_16x16x32_bf16(
                false, afrag[s].v, false, b0.v, (short)0, acc0, false, false);
            acc1 = __builtin_amdgcn_wmma_f32_16x16x32_bf16(
                false, afrag[s].v, false, b1.v, (short)0, acc1, false, false);
        }
        // fused epilogue for both 16x16 tiles (branch-free tanh)
        int n0 = j * 16 + lr, n1 = n0 + 16;
        float w1b0 = W1b[n0], lp0 = locproj[n0], vw0 = Vw[n0];
        float w1b1 = W1b[n1], lp1 = locproj[n1], vw1 = Vw[n1];
#pragma unroll
        for (int idx = 0; idx < 8; idx++) {
            float e0 = acc0[idx] + w1b0 + qproj[qb[idx] + n0] + cvv[idx] * lp0;
            float e1 = acc1[idx] + w1b1 + qproj[qb[idx] + n1] + cvv[idx] * lp1;
            sacc[idx] += fast_tanh(e0) * vw0 + fast_tanh(e1) * vw1;
        }
    }

    // reduce over the 16 columns held by the 16 lanes of each half-wave
#pragma unroll
    for (int idx = 0; idx < 8; idx++) {
        float v = sacc[idx];
        v += __shfl_xor(v, 1, 32);
        v += __shfl_xor(v, 2, 32);
        v += __shfl_xor(v, 4, 32);
        v += __shfl_xor(v, 8, 32);
        sacc[idx] = v;
    }
    if (lr == 0) {
        float vb = Vb[0];
#pragma unroll
        for (int idx = 0; idx < 8; idx++)
            rawScore[mrow[idx]] = sacc[idx] + vb;
    }
}

// ----------------------------------- k4: per-batch top-K mask (radix select)
__global__ void topk_kernel(const float* __restrict__ raw, float* __restrict__ outScore) {
    __shared__ float    ss[T_];
    __shared__ unsigned su[T_];
    __shared__ int      red[256];
    __shared__ int      scanbuf[256];
    const int b = blockIdx.x, tid = threadIdx.x;

    for (int i = tid; i < T_; i += 256) {
        float s = raw[b * T_ + i];
        ss[i] = s;
        unsigned ub = __float_as_uint(s);
        su[i] = (ub & 0x80000000u) ? ~ub : (ub | 0x80000000u);  // order-preserving
    }
    __syncthreads();

    // binary search: largest x with count(su >= x) >= K  => K-th largest value
    unsigned lo = 0u;
    for (int bit = 31; bit >= 0; bit--) {
        unsigned cand = lo | (1u << bit);
        int c = 0;
        for (int i = tid; i < T_; i += 256) c += (su[i] >= cand);
        red[tid] = c; __syncthreads();
        for (int off = 128; off > 0; off >>= 1) {
            if (tid < off) red[tid] += red[tid + off];
            __syncthreads();
        }
        int cnt = red[0]; __syncthreads();
        if (cnt >= K_) lo = cand;
    }
    const unsigned uT = lo;

    int c = 0;
    for (int i = tid; i < T_; i += 256) c += (su[i] > uT);
    red[tid] = c; __syncthreads();
    for (int off = 128; off > 0; off >>= 1) {
        if (tid < off) red[tid] += red[tid + off];
        __syncthreads();
    }
    const int nGreater = red[0]; __syncthreads();
    const int nTiesKeep = K_ - nGreater;           // top_k keeps lowest-index ties

    // tie ranks via contiguous chunks + block scan
    const int i0 = tid * 8;
    int cntT = 0;
    for (int i = i0; i < i0 + 8 && i < T_; i++) cntT += (su[i] == uT);
    scanbuf[tid] = cntT; __syncthreads();
    for (int off = 1; off < 256; off <<= 1) {
        int v = (tid >= off) ? scanbuf[tid - off] : 0;
        __syncthreads();
        scanbuf[tid] += v;
        __syncthreads();
    }
    int rank = scanbuf[tid] - cntT;                // exclusive prefix
    for (int i = i0; i < i0 + 8 && i < T_; i++) {
        unsigned u = su[i];
        bool keep;
        if (u > uT)       keep = true;
        else if (u == uT) { keep = (rank < nTiesKeep); rank++; }
        else              keep = false;
        outScore[b * T_ + i] = keep ? ss[i] : 0.0f;
    }
}

// ------------------------------- k5: denom[t] = sum_b sigmoid(score[b,t])
__global__ void denom_kernel(const float* __restrict__ score, float* __restrict__ denom) {
    int t = blockIdx.x * 256 + threadIdx.x;
    if (t >= T_) return;
    float d = 0.f;
    for (int b = 0; b < B_; b++)                    // coalesced across t
        d += fast_sigmoid(score[b * T_ + t]);
    denom[t] = d;
}

// ------------------------------- k6: attention weights
__global__ void aw_kernel(const float* __restrict__ score,
                          const float* __restrict__ denom,
                          float* __restrict__ aw) {
    int gid = blockIdx.x * 256 + threadIdx.x;
    if (gid >= MTOT) return;
    int t = gid % T_;
    aw[gid] = fast_sigmoid(score[gid]) * __builtin_amdgcn_rcpf(denom[t]) *
              denom[t] / denom[t];                  // keep exact div semantics? no:
}
// NOTE: simpler exact form used below instead (see aw_kernel2)
__global__ void aw_kernel2(const float* __restrict__ score,
                           const float* __restrict__ denom,
                           float* __restrict__ aw) {
    int gid = blockIdx.x * 256 + threadIdx.x;
    if (gid >= MTOT) return;
    int t = gid % T_;
    aw[gid] = fast_sigmoid(score[gid]) / denom[t];
}

// ------------------------------- k7/k8: context vector (deterministic split)
__global__ void ctx_partial_kernel(const float* __restrict__ aw,
                                   const float* __restrict__ values,
                                   float* __restrict__ part) {
    const int b  = blockIdx.x >> 2;                // TSPLIT = 4
    const int sp = blockIdx.x & 3;
    const int h  = threadIdx.x;
    const int t0 = sp * 500;
    const int t1 = min(t0 + 500, T_);
    float acc = 0.f;
    for (int t = t0; t < t1; t++) {
        float a = aw[b * T_ + t];                  // wave-uniform
        acc += a * values[((size_t)(b * T_ + t)) * H_ + h];  // coalesced
    }
    part[blockIdx.x * H_ + h] = acc;
}

__global__ void ctx_reduce_kernel(const float* __restrict__ part, float* __restrict__ ctx) {
    const int b = blockIdx.x, h = threadIdx.x;
    float acc = 0.f;
    for (int sp = 0; sp < 4; sp++) acc += part[(b * 4 + sp) * H_ + h];
    ctx[b * H_ + h] = acc;
}

// ---------------------------------------------------------------- launcher
extern "C" void kernel_launch(void* const* d_in, const int* in_sizes, int n_in,
                              void* d_out, int out_size, void* d_ws, size_t ws_size,
                              hipStream_t stream) {
    const float* query   = (const float*)d_in[0];
    const float* values  = (const float*)d_in[1];
    const float* prevatt = (const float*)d_in[2];
    const float* W1w     = (const float*)d_in[3];
    const float* W1b     = (const float*)d_in[4];
    const float* W2w     = (const float*)d_in[5];
    const float* W2b     = (const float*)d_in[6];
    const float* Vw      = (const float*)d_in[7];
    const float* Vb      = (const float*)d_in[8];
    const float* convw   = (const float*)d_in[9];
    const float* locproj = (const float*)d_in[10];

    float* out   = (float*)d_out;
    float* ctx   = out;                      // [B,H]
    float* aw    = out + B_ * H_;            // [B,T]
    float* score = aw + MTOT;                // [B,T] (masked)

    // workspace layout (floats)
    float* ws     = (float*)d_ws;
    float* paT    = ws;                       // T*B
    float* convoT = paT + (size_t)T_ * B_;    // T*B
    float* qproj  = convoT + (size_t)T_ * B_; // B*U
    float* raw    = qproj + (size_t)B_ * U_;  // B*T
    float* denom  = raw + (size_t)MTOT;       // T
    float* part   = denom + 2048;             // B*4*H

    qproj_kernel<<<B_, 256, 0, stream>>>(query, W2w, W2b, qproj);
    transpose_pa<<<(MTOT + 255) / 256, 256, 0, stream>>>(prevatt, paT);
    conv_kernel<<<(T_ + 15) / 16, 128, 0, stream>>>(paT, convw, convoT);

    const size_t shmem = (size_t)(256 * 264 + 128 * 264) * 2;  // ~198.4 KB < 320 KB
    score_gemm_kernel<<<MTOT / 128, 256, shmem, stream>>>(
        values, W1w, W1b, qproj, convoT, locproj, Vw, Vb, raw);

    topk_kernel<<<B_, 256, 0, stream>>>(raw, score);
    denom_kernel<<<(T_ + 255) / 256, 256, 0, stream>>>(score, denom);
    aw_kernel2<<<(MTOT + 255) / 256, 256, 0, stream>>>(score, denom, aw);
    ctx_partial_kernel<<<B_ * 4, 256, 0, stream>>>(aw, values, part);
    ctx_reduce_kernel<<<B_, 256, 0, stream>>>(part, ctx);
}
